// MultiLayerGNN_26757646254600
// MI455X (gfx1250) — matrix operand order repaired and checked by reading, not compiled, not run on hardware
//
#include <hip/hip_runtime.h>

// ---------------------------------------------------------------------------
// MultiLayerGNN on MI455X (gfx1250): bf16 WMMA + async global->LDS copies.
// adj_norm converted once to bf16 (128MB); dense feature operands are
// pre-transposed so every GEMM is C = A * B^T with K-contiguous rows ->
// all LDS tile fills are aligned global_load_async_to_lds_b128 (no VGPR
// staging, no LDS scatter), overlapped with v_wmma_f32_16x16x32_bf16.
// ---------------------------------------------------------------------------

#define NN 8192
#define EE 256

typedef __attribute__((ext_vector_type(16))) __bf16 v16bf;
typedef __attribute__((ext_vector_type(8)))  float  v8f;

__device__ __forceinline__ unsigned short f2bf(float f) {
    unsigned u = __float_as_uint(f);
    u += 0x7FFFu + ((u >> 16) & 1u);       // round-to-nearest-even
    return (unsigned short)(u >> 16);
}

// --------------------------- degree / rsqrt --------------------------------
__global__ __launch_bounds__(256) void deg_kernel(const float* __restrict__ adj,
                                                  float* __restrict__ d) {
    __shared__ float red[256];
    const int row = blockIdx.x;
    const float4* a4 = (const float4*)(adj + (size_t)row * NN);
    float s = 0.f;
    for (int j = threadIdx.x; j < NN / 4; j += 256) {
        float4 v = a4[j];
        s += v.x + v.y + v.z + v.w;
    }
    red[threadIdx.x] = s; __syncthreads();
    for (int o = 128; o > 0; o >>= 1) {
        if (threadIdx.x < o) red[threadIdx.x] += red[threadIdx.x + o];
        __syncthreads();
    }
    if (threadIdx.x == 0) d[row] = rsqrtf(fmaxf(red[0], 1e-6f));
}

// --------------- adj_norm = d[i]*adj*d[j], stored bf16 ---------------------
__global__ __launch_bounds__(256) void norm_adj_kernel(const float* __restrict__ adj,
                                                       const float* __restrict__ d,
                                                       unsigned short* __restrict__ adjn) {
    const size_t total4 = (size_t)NN * NN / 4;
    for (size_t idx = (size_t)blockIdx.x * blockDim.x + threadIdx.x; idx < total4;
         idx += (size_t)gridDim.x * blockDim.x) {
        size_t i = idx / (NN / 4);
        size_t j = (idx % (NN / 4)) * 4;
        float4 a = ((const float4*)adj)[idx];
        float di = d[i];
        unsigned p0 = (unsigned)f2bf(di * a.x * d[j])     | ((unsigned)f2bf(di * a.y * d[j + 1]) << 16);
        unsigned p1 = (unsigned)f2bf(di * a.z * d[j + 2]) | ((unsigned)f2bf(di * a.w * d[j + 3]) << 16);
        ((uint2*)adjn)[idx] = make_uint2(p0, p1);
    }
}

// --------------------------- f32 -> bf16 (weights) -------------------------
__global__ __launch_bounds__(256) void cvt_bf16_kernel(const float* __restrict__ in,
                                                       unsigned short* __restrict__ out, int n) {
    for (int i = blockIdx.x * blockDim.x + threadIdx.x; i < n; i += gridDim.x * blockDim.x)
        out[i] = f2bf(in[i]);
}

// ------------- 64x64 LDS-tiled transpose: f32[R][C] -> bf16[C][R] ----------
__global__ __launch_bounds__(256) void transpose_f32_bf16_kernel(
    const float* __restrict__ in, unsigned short* __restrict__ out, int R, int C) {
    __shared__ unsigned short t[64][65];
    const int c0 = blockIdx.x * 64, r0 = blockIdx.y * 64;
    const int tc = (threadIdx.x & 15) * 4, tr = threadIdx.x >> 4;
#pragma unroll
    for (int i = 0; i < 4; ++i) {
        int r = tr + i * 16;
        float4 v = *(const float4*)(in + (size_t)(r0 + r) * C + c0 + tc);
        t[tc + 0][r] = f2bf(v.x); t[tc + 1][r] = f2bf(v.y);
        t[tc + 2][r] = f2bf(v.z); t[tc + 3][r] = f2bf(v.w);
    }
    __syncthreads();
    const int orr = (threadIdx.x & 15) * 4, oc = threadIdx.x >> 4;
#pragma unroll
    for (int i = 0; i < 4; ++i) {
        int c = oc + i * 16;
        unsigned p0 = (unsigned)t[c][orr + 0] | ((unsigned)t[c][orr + 1] << 16);
        unsigned p1 = (unsigned)t[c][orr + 2] | ((unsigned)t[c][orr + 3] << 16);
        *(uint2*)(out + (size_t)(c0 + c) * R + r0 + orr) = make_uint2(p0, p1);
    }
}

// ------------- 64x64 LDS-tiled transpose: bf16[R][C] -> bf16[C][R] ---------
__global__ __launch_bounds__(256) void transpose_bf16_kernel(
    const unsigned short* __restrict__ in, unsigned short* __restrict__ out, int R, int C) {
    __shared__ unsigned short t[64][65];
    const int c0 = blockIdx.x * 64, r0 = blockIdx.y * 64;
    const int tc = (threadIdx.x & 15) * 4, tr = threadIdx.x >> 4;
#pragma unroll
    for (int i = 0; i < 4; ++i) {
        int r = tr + i * 16;
        uint2 v = *(const uint2*)(in + (size_t)(r0 + r) * C + c0 + tc);
        t[tc + 0][r] = (unsigned short)(v.x & 0xFFFF);
        t[tc + 1][r] = (unsigned short)(v.x >> 16);
        t[tc + 2][r] = (unsigned short)(v.y & 0xFFFF);
        t[tc + 3][r] = (unsigned short)(v.y >> 16);
    }
    __syncthreads();
    const int orr = (threadIdx.x & 15) * 4, oc = threadIdx.x >> 4;
#pragma unroll
    for (int i = 0; i < 4; ++i) {
        int c = oc + i * 16;
        unsigned p0 = (unsigned)t[c][orr + 0] | ((unsigned)t[c][orr + 1] << 16);
        unsigned p1 = (unsigned)t[c][orr + 2] | ((unsigned)t[c][orr + 3] << 16);
        *(uint2*)(out + (size_t)(c0 + c) * R + r0 + orr) = make_uint2(p0, p1);
    }
}

// ---------------------------------------------------------------------------
// bf16 WMMA GEMM:  C[M,N] = A[M,K] * B[N,K]^T      (both K-contiguous)
// BM=128 BN=128 BK=32; 256 threads = 8 waves (2x4); each wave: 4x2 16x16 tiles.
// Global->LDS tile fills use gfx1250 async copies (ASYNCcnt), double-buffered
// against the WMMA stage. LDS rows padded to 40 bf16 (80B: 16B-aligned,
// bank-rotating) so every fragment load is a pair of ds_load_b128.
// ---------------------------------------------------------------------------
template <bool OUT_BF16>
__global__ __launch_bounds__(256) void gemm_bf16_wmma(const unsigned short* __restrict__ A, int lda,
                                                      const unsigned short* __restrict__ B, int ldb,
                                                      void* __restrict__ Cout, int ldc, int K) {
    const int BK = 32, LDT = 40;
    __shared__ unsigned short As[2][128 * 40];
    __shared__ unsigned short Bs[2][128 * 40];

    const int tid   = threadIdx.x;
    const int lane  = tid & 31;
    const int wave  = tid >> 5;
    const int waveM = wave & 1;   // 2 wave rows -> 64 M each
    const int waveN = wave >> 1;  // 4 wave cols -> 32 N each
    const int half  = lane >> 4;
    const int ln16  = lane & 15;

    const int m0 = blockIdx.y * 128;
    const int n0 = blockIdx.x * 128;

    v8f acc[4][2] = {};

    // Async global->LDS: each thread issues 2 A + 2 B b128 copies per stage.
    auto issueAsync = [&](int kt, int s) {
#pragma unroll
        for (int i = 0; i < 2; ++i) {
            int idx = tid + i * 256;            // 512 vectors of 8 bf16 per tile
            int row = idx >> 2, kv = idx & 3;
            const unsigned short* ga = A + (size_t)(m0 + row) * lda + kt * BK + kv * 8;
            unsigned la = (unsigned)(size_t)&As[s][row * LDT + kv * 8];
            asm volatile("global_load_async_to_lds_b128 %0, %1, off"
                         :: "v"(la), "v"(ga) : "memory");
            const unsigned short* gb = B + (size_t)(n0 + row) * ldb + kt * BK + kv * 8;
            unsigned lb = (unsigned)(size_t)&Bs[s][row * LDT + kv * 8];
            asm volatile("global_load_async_to_lds_b128 %0, %1, off"
                         :: "v"(lb), "v"(gb) : "memory");
        }
    };
    auto waitAsync = [&]() {
        asm volatile("s_wait_asynccnt 0x0" ::: "memory");
    };
    auto compute = [&](int s) {
        union Frag { uint4 u[2]; v16bf v; } af[4], bfr[2];
#pragma unroll
        for (int tm = 0; tm < 4; ++tm) {        // A 16x32: pair-interleaved halves
            const unsigned short* p = &As[s][(waveM * 64 + tm * 16 + ln16) * LDT];
            af[tm].u[0] = *(const uint4*)(p + 8 * half);
            af[tm].u[1] = *(const uint4*)(p + 16 + 8 * half);
        }
#pragma unroll
        for (int tn = 0; tn < 2; ++tn) {        // B 32x16: pair-interleaved halves
            const unsigned short* p = &Bs[s][(waveN * 32 + tn * 16 + ln16) * LDT];
            bfr[tn].u[0] = *(const uint4*)(p + 8 * half);
            bfr[tn].u[1] = *(const uint4*)(p + 16 + 8 * half);
        }
#pragma unroll
        for (int tm = 0; tm < 4; ++tm)
#pragma unroll
            for (int tn = 0; tn < 2; ++tn)
                acc[tm][tn] = __builtin_amdgcn_wmma_f32_16x16x32_bf16(
                    false, af[tm].v, false, bfr[tn].v, (short)0, acc[tm][tn], false, false);
    };

    const int nk = K / BK;
    issueAsync(0, 0);
    waitAsync();
    __syncthreads();
    int s = 0;
    for (int kt = 0; kt < nk; ++kt) {
        if (kt + 1 < nk) issueAsync(kt + 1, s ^ 1);   // overlap copy with WMMA
        compute(s);
        if (kt + 1 < nk) {
            waitAsync();
            __syncthreads();
            s ^= 1;
        }
    }

    // D layout: lane n = lane&15, m = r + 8*(lane>>4)
#pragma unroll
    for (int tm = 0; tm < 4; ++tm)
#pragma unroll
        for (int tn = 0; tn < 2; ++tn) {
            int mbase = m0 + waveM * 64 + tm * 16 + 8 * half;
            int n     = n0 + waveN * 32 + tn * 16 + ln16;
#pragma unroll
            for (int r = 0; r < 8; ++r) {
                float v = acc[tm][tn][r];
                if (OUT_BF16)
                    ((unsigned short*)Cout)[(size_t)(mbase + r) * ldc + n] = f2bf(v);
                else
                    ((float*)Cout)[(size_t)(mbase + r) * ldc + n] = v;
            }
        }
}

// ---------------- bias + LayerNorm + exact GELU (+ residual) ---------------
template <bool FINAL>
__global__ __launch_bounds__(256) void ln_gelu_kernel(const float* __restrict__ h,
                                                      const float* __restrict__ bias,
                                                      const float* __restrict__ g,
                                                      const float* __restrict__ beta,
                                                      unsigned short* __restrict__ out_bf,
                                                      const float* __restrict__ resid,
                                                      float* __restrict__ out_f) {
    __shared__ float red[256];
    const int row = blockIdx.x, tid = threadIdx.x;
    const size_t off = (size_t)row * EE + tid;
    float v = h[off] + bias[tid];              // bias shifts the mean -> pre-LN
    red[tid] = v; __syncthreads();
    for (int o = 128; o > 0; o >>= 1) { if (tid < o) red[tid] += red[tid + o]; __syncthreads(); }
    float mu = red[0] * (1.0f / EE);
    __syncthreads();
    float dv = v - mu;
    red[tid] = dv * dv; __syncthreads();
    for (int o = 128; o > 0; o >>= 1) { if (tid < o) red[tid] += red[tid + o]; __syncthreads(); }
    float var = red[0] * (1.0f / EE);
    float xn = dv * rsqrtf(var + 1e-5f) * g[tid] + beta[tid];
    float ge = 0.5f * xn * (1.0f + erff(xn * 0.70710678118654752f));  // exact GELU
    if (FINAL) out_f[off] = ge + resid[off];
    else       out_bf[off] = f2bf(ge);
}

// ---------------------------------------------------------------------------
extern "C" void kernel_launch(void* const* d_in, const int* in_sizes, int n_in,
                              void* d_out, int out_size, void* d_ws, size_t ws_size,
                              hipStream_t stream) {
    (void)in_sizes; (void)n_in; (void)out_size; (void)ws_size;
    const float* x   = (const float*)d_in[0];
    const float* adj = (const float*)d_in[1];
    const float* W1  = (const float*)d_in[2];
    const float* b1  = (const float*)d_in[3];
    const float* g1  = (const float*)d_in[4];
    const float* be1 = (const float*)d_in[5];
    const float* W2  = (const float*)d_in[6];
    const float* b2  = (const float*)d_in[7];
    const float* g2  = (const float*)d_in[8];
    const float* be2 = (const float*)d_in[9];
    float* out = (float*)d_out;

    char* ws = (char*)d_ws;
    size_t o = 0;
    auto take = [&](size_t bytes) -> char* {
        char* p = ws + o;
        o = (o + bytes + 255) & ~(size_t)255;
        return p;
    };
    float*          dS   = (float*)take((size_t)NN * 4);               //  32 KB
    unsigned short* adjn = (unsigned short*)take((size_t)NN * NN * 2); // 128 MB
    unsigned short* xbT  = (unsigned short*)take((size_t)NN * EE * 2); //   4 MB  (EE x NN)
    unsigned short* w1b  = (unsigned short*)take((size_t)EE * EE * 2);
    unsigned short* w2b  = (unsigned short*)take((size_t)EE * EE * 2);
    unsigned short* aggb = (unsigned short*)take((size_t)NN * EE * 2); //   4 MB
    float*          hf   = (float*)take((size_t)NN * EE * 4);          //   8 MB
    unsigned short* hb   = (unsigned short*)take((size_t)NN * EE * 2); //   4 MB
    unsigned short* hbT  = (unsigned short*)take((size_t)NN * EE * 2); //   4 MB  (EE x NN)

    deg_kernel<<<NN, 256, 0, stream>>>(adj, dS);
    norm_adj_kernel<<<8192, 256, 0, stream>>>(adj, dS, adjn);
    transpose_f32_bf16_kernel<<<dim3(EE / 64, NN / 64), 256, 0, stream>>>(x, xbT, NN, EE);
    cvt_bf16_kernel<<<64, 256, 0, stream>>>(W1, w1b, EE * EE);
    cvt_bf16_kernel<<<64, 256, 0, stream>>>(W2, w2b, EE * EE);

    dim3 grid(EE / 128, NN / 128);  // (2, 64)
    // layer 1: aggregate -> linear(W1^T) -> LN -> GELU
    gemm_bf16_wmma<true ><<<grid, 256, 0, stream>>>(adjn, NN, xbT, NN, aggb, EE, NN);
    gemm_bf16_wmma<false><<<grid, 256, 0, stream>>>(aggb, EE, w1b, EE, hf,   EE, EE);
    ln_gelu_kernel<false><<<NN, 256, 0, stream>>>(hf, b1, g1, be1, hb, nullptr, nullptr);
    transpose_bf16_kernel<<<dim3(EE / 64, NN / 64), 256, 0, stream>>>(hb, hbT, NN, EE);
    // layer 2 + residual
    gemm_bf16_wmma<true ><<<grid, 256, 0, stream>>>(adjn, NN, hbT, NN, aggb, EE, NN);
    gemm_bf16_wmma<false><<<grid, 256, 0, stream>>>(aggb, EE, w2b, EE, hf,   EE, EE);
    ln_gelu_kernel<true><<<NN, 256, 0, stream>>>(hf, b2, g2, be2, nullptr, x, out);
}
